// Res18_RFM_19061064859921
// MI455X (gfx1250) — matrix operand order repaired and checked by reading, not compiled, not run on hardware
//
#include <hip/hip_runtime.h>
#include <math.h>

// ---------------------------------------------------------------------------
// Types
// ---------------------------------------------------------------------------
typedef _Float16 h16;
typedef __attribute__((ext_vector_type(16))) _Float16 v16h;
typedef __attribute__((ext_vector_type(8)))  float    v8f;
typedef __attribute__((ext_vector_type(4)))  unsigned u32x4;
typedef __attribute__((ext_vector_type(8)))  int      i32x8;
typedef __attribute__((ext_vector_type(4)))  int      i32x4;

#if defined(__has_builtin)
#if __has_builtin(__builtin_amdgcn_tensor_load_to_lds) && __has_builtin(__builtin_amdgcn_s_wait_tensorcnt)
#define USE_TDM 1
#endif
#endif
#ifndef USE_TDM
#define USE_TDM 0
#endif

// Low 32 bits of a generic pointer to LDS == LDS byte offset (flat->LDS drops
// the upper bits per CDNA5 aperture rules).
static __device__ __forceinline__ unsigned ldsAddr(const void* p) {
  return (unsigned)(unsigned long long)p;
}

#if USE_TDM
// 2-D TDM tile load: (rows x 32) f16 tile, row pitch = rowElems (f16 units).
static __device__ __forceinline__ void tdm_load_tile(unsigned ldsOff, const void* g,
                                                     unsigned rowElems, unsigned rows) {
  unsigned long long ga = (unsigned long long)g;
  u32x4 g0;
  g0[0] = 1u;                                   // count=1, user descriptor
  g0[1] = ldsOff;                               // lds_addr
  g0[2] = (unsigned)ga;                         // global_addr[31:0]
  g0[3] = (unsigned)((ga >> 32) & 0x1FFFFFFull) | (2u << 30);   // addr[56:32] | type=2
  i32x8 g1;
  g1[0] = (int)(1u << 16);                      // data_size=1 (2 bytes)
  g1[1] = (int)((rowElems & 0xFFFFu) << 16);    // tensor_dim0 lo16
  g1[2] = (int)(rowElems >> 16);                // tensor_dim0 hi | tensor_dim1 lo16 (=0)
  g1[3] = (int)(((1u << 20) >> 16) | (32u << 16)); // tensor_dim1 hi (2^20), tile_dim0=32
  g1[4] = (int)rows;                            // tile_dim1=rows, tile_dim2=0
  g1[5] = (int)rowElems;                        // tensor_dim0_stride lo32
  g1[6] = 0;                                    // stride hi, dim1_stride (unused)
  g1[7] = 0;
  i32x4 g2 = {0, 0, 0, 0}, g3 = {0, 0, 0, 0};
#if defined(__clang_major__) && __clang_major__ >= 23
  i32x8 g4 = {0, 0, 0, 0, 0, 0, 0, 0};
  __builtin_amdgcn_tensor_load_to_lds(g0, g1, g2, g3, g4, 0);
#else
  __builtin_amdgcn_tensor_load_to_lds(g0, g1, g2, g3, 0);
#endif
}
#endif

// ---------------------------------------------------------------------------
// WMMA GEMM: C(MxN) = act(A(MxK) * BT(NxK)^T + bias), f16 in, f32 accum.
// 128 threads (4 waves). 64x64 macro tile; each wave owns a 32x32 quadrant
// (2x2 WMMA 16x16 tiles -> 4 v_wmma per k-step, fragments reused 2x).
// Double-buffered LDS tiles; TDM prefetches the next K-step while the
// current one is multiplied (wait tensorcnt<=2 = previous pair complete).
// M, N multiples of 64; K multiple of 32 (guaranteed by call sites).
// ---------------------------------------------------------------------------
__global__ __launch_bounds__(128) void k_gemm(const h16* __restrict__ A,
                                              const h16* __restrict__ BT, int M, int N,
                                              int K, const float* __restrict__ bias,
                                              int act, float* __restrict__ Cf,
                                              h16* __restrict__ Ch, int ldc) {
  __shared__ __align__(16) h16 As[2][64 * 32];
  __shared__ __align__(16) h16 Bs[2][64 * 32];
  const int m0 = blockIdx.y * 64, n0 = blockIdx.x * 64;
  const int lane = threadIdx.x & 31, wave = threadIdx.x >> 5;
  const int mi = wave >> 1, ni = wave & 1;   // 2x2 wave grid over the 64x64 tile
  const int nk = K >> 5;
  v8f acc00 = {0.f,0.f,0.f,0.f,0.f,0.f,0.f,0.f};
  v8f acc01 = acc00, acc10 = acc00, acc11 = acc00;

#if USE_TDM
  if (wave == 0) {  // prologue: fetch k-step 0 into buffer 0
    tdm_load_tile(ldsAddr(As[0]), (const char*)A + (size_t)m0 * K * 2, (unsigned)K, 64u);
    tdm_load_tile(ldsAddr(Bs[0]), (const char*)BT + (size_t)n0 * K * 2, (unsigned)K, 64u);
  }
#endif
  for (int ki = 0; ki < nk; ++ki) {
    const int p = ki & 1;
#if USE_TDM
    if (wave == 0) {
      if (ki + 1 < nk) {  // prefetch next k-step into the other buffer
        const size_t kb = (size_t)(ki + 1) * 32 * 2;
        tdm_load_tile(ldsAddr(As[p ^ 1]), (const char*)A + (size_t)m0 * K * 2 + kb,
                      (unsigned)K, 64u);
        tdm_load_tile(ldsAddr(Bs[p ^ 1]), (const char*)BT + (size_t)n0 * K * 2 + kb,
                      (unsigned)K, 64u);
        __builtin_amdgcn_s_wait_tensorcnt(2);  // pair for buffer p is complete
      } else {
        __builtin_amdgcn_s_wait_tensorcnt(0);
      }
    }
#else
    for (int i = threadIdx.x; i < 2048; i += 128) {
      int r = i >> 5, cc = i & 31;
      As[p][i] = A[(size_t)(m0 + r) * K + ki * 32 + cc];
      Bs[p][i] = BT[(size_t)(n0 + r) * K + ki * 32 + cc];
    }
#endif
    __syncthreads();
    union F { unsigned u[8]; v16h v; };
    F a0, a1, b0, b1;
    const int ha = (lane >> 4) * 4;   // A: K-half base (uints)
    const int hb = (lane >> 4) * 4;   // B: K-half base (uints)
    {
      const unsigned* ap0 = (const unsigned*)(&As[p][(size_t)(32 * mi + (lane & 15)) * 32]);
      const unsigned* ap1 = (const unsigned*)(&As[p][(size_t)(32 * mi + 16 + (lane & 15)) * 32]);
      const unsigned* bp0 = (const unsigned*)(&Bs[p][(size_t)(32 * ni + (lane & 15)) * 32]);
      const unsigned* bp1 = (const unsigned*)(&Bs[p][(size_t)(32 * ni + 16 + (lane & 15)) * 32]);
#pragma unroll
      for (int j = 0; j < 4; ++j) {
        a0.u[j] = ap0[ha + j]; a0.u[4 + j] = ap0[8 + ha + j];
        a1.u[j] = ap1[ha + j]; a1.u[4 + j] = ap1[8 + ha + j];
      }
#pragma unroll
      for (int j = 0; j < 4; ++j) { b0.u[j] = bp0[hb + j]; b0.u[4 + j] = bp0[8 + hb + j]; b1.u[j] = bp1[hb + j]; b1.u[4 + j] = bp1[8 + hb + j]; }
    }
    acc00 = __builtin_amdgcn_wmma_f32_16x16x32_f16(false, a0.v, false, b0.v, (short)0, acc00, false, false);
    acc01 = __builtin_amdgcn_wmma_f32_16x16x32_f16(false, a0.v, false, b1.v, (short)0, acc01, false, false);
    acc10 = __builtin_amdgcn_wmma_f32_16x16x32_f16(false, a1.v, false, b0.v, (short)0, acc10, false, false);
    acc11 = __builtin_amdgcn_wmma_f32_16x16x32_f16(false, a1.v, false, b1.v, (short)0, acc11, false, false);
    __syncthreads();
  }
  // Epilogue: bias + activation + store (f32 or f16)
  const int nb = n0 + 32 * ni + (lane & 15);
  const int mb = m0 + 32 * mi + ((lane >> 4) ? 8 : 0);
#pragma unroll
  for (int r = 0; r < 2; ++r) {
#pragma unroll
    for (int s = 0; s < 2; ++s) {
      const v8f* cc = (r == 0) ? (s == 0 ? &acc00 : &acc01) : (s == 0 ? &acc10 : &acc11);
      int n = nb + 16 * s;
#pragma unroll
      for (int i = 0; i < 8; ++i) {
        int m = mb + 16 * r + i;
        float v = (*cc)[i];
        if (bias) v += bias[m];
        if (act == 1) v = fmaxf(v, 0.f);
        else if (act == 2) v = tanhf(v);
        if (Cf) Cf[(size_t)m * ldc + n] = v;
        else    Ch[(size_t)m * ldc + n] = (h16)v;
      }
    }
  }
}

// ---------------------------------------------------------------------------
// im2col (writes the column buffer TRANSPOSED: col[n][Kpad], K-contiguous,
// so it feeds k_gemm's BT operand directly). stride 2, SAME pad = (0,1).
// ---------------------------------------------------------------------------
__global__ void k_im2col_f32(h16* __restrict__ col, const float* __restrict__ src,
                             int Cin, int H, int W, int Ho, int Wo, int Kpad) {
  int idx = blockIdx.x * blockDim.x + threadIdx.x;
  int total = Ho * Wo * Kpad;
  if (idx >= total) return;
  int n = idx / Kpad, k = idx % Kpad;
  float v = 0.f;
  if (k < Cin * 9) {
    int ci = k / 9, r = k % 9, ky = r / 3, kx = r % 3;
    int oy = n / Wo, ox = n % Wo;
    int iy = 2 * oy + ky, ix = 2 * ox + kx;
    if (iy < H && ix < W) v = src[(size_t)(ci * H + iy) * W + ix];
  }
  col[idx] = (h16)v;
}

__global__ void k_im2col_h(h16* __restrict__ col, const h16* __restrict__ src, int Cin,
                           int H, int W, int Ho, int Wo, int Kpad) {
  int idx = blockIdx.x * blockDim.x + threadIdx.x;
  int total = Ho * Wo * Kpad;
  if (idx >= total) return;
  int n = idx / Kpad, k = idx % Kpad;
  float v = 0.f;
  if (k < Cin * 9) {
    int ci = k / 9, r = k % 9, ky = r / 3, kx = r % 3;
    int oy = n / Wo, ox = n % Wo;
    int iy = 2 * oy + ky, ix = 2 * ox + kx;
    if (iy < H && ix < W) v = (float)src[(size_t)(ci * H + iy) * W + ix];
  }
  col[idx] = (h16)v;
}

// ---------------------------------------------------------------------------
// Weight prep / transposes
// ---------------------------------------------------------------------------
__global__ void k_w1pack(h16* dst, const float* w) {  // (64,9) -> (64,32) zero padded
  int i = blockIdx.x * blockDim.x + threadIdx.x;
  if (i >= 64 * 32) return;
  int co = i >> 5, k = i & 31;
  dst[i] = (k < 9) ? (h16)w[co * 9 + k] : (h16)0.f;
}
__global__ void k_cvt_h(h16* dst, const float* src, int n) {
  int i = blockIdx.x * blockDim.x + threadIdx.x;
  if (i < n) dst[i] = (h16)src[i];
}
__global__ void k_zero_h(h16* dst, int n) {
  int i = blockIdx.x * blockDim.x + threadIdx.x;
  if (i < n) dst[i] = (h16)0.f;
}
// dst[r][c] = src[c][r]; dst is rows x cols row-major (f16), src is cols x rows (f32)
__global__ void k_transpose_f32h(h16* dst, const float* src, int rows, int cols) {
  int i = blockIdx.x * blockDim.x + threadIdx.x;
  if (i >= rows * cols) return;
  int r = i / cols, c = i % cols;
  dst[i] = (h16)src[(size_t)c * rows + r];
}
__global__ void k_transpose_h(h16* dst, const h16* src, int rows, int cols) {
  int i = blockIdx.x * blockDim.x + threadIdx.x;
  if (i >= rows * cols) return;
  int r = i / cols, c = i % cols;
  dst[i] = src[(size_t)c * rows + r];
}

// Pack relu sign bits of h1 (per-batch) so h1 need not be kept resident.
__global__ void k_mask1(unsigned* bits, const h16* h, int nwords) {
  int i = blockIdx.x * blockDim.x + threadIdx.x;
  if (i >= nwords) return;
  unsigned w = 0;
  int base = i * 32;
#pragma unroll
  for (int j = 0; j < 32; ++j)
    if ((float)h[base + j] > 0.f) w |= 1u << j;
  bits[i] = w;
}

// ---------------------------------------------------------------------------
// Attention + statistics pooling
// ---------------------------------------------------------------------------
__global__ void k_score(float* s, const h16* att, const float* va) {  // s[b,t]=va.att
  int i = blockIdx.x * blockDim.x + threadIdx.x;
  if (i >= 32 * 128) return;
  int b = i >> 7, t = i & 127;
  float acc = 0.f;
  const h16* a = att + (size_t)b * 16384 + t;
  for (int h = 0; h < 128; ++h) acc += va[h] * (float)a[h * 128];
  s[i] = acc;
}
__global__ void k_softmax(float* alpha, const float* s) {  // over T=128 per batch
  int b = blockIdx.x, t = threadIdx.x;
  __shared__ float sm[128];
  float v = s[b * 128 + t];
  sm[t] = v; __syncthreads();
  for (int w = 64; w; w >>= 1) { if (t < w) sm[t] = fmaxf(sm[t], sm[t + w]); __syncthreads(); }
  float mx = sm[0]; __syncthreads();
  float e = expf(v - mx);
  sm[t] = e; __syncthreads();
  for (int w = 64; w; w >>= 1) { if (t < w) sm[t] += sm[t + w]; __syncthreads(); }
  alpha[b * 128 + t] = e / sm[0];
}
__global__ void k_pool(float* mu, float* m2, const float* alpha, const h16* feat) {
  int i = blockIdx.x * blockDim.x + threadIdx.x;
  if (i >= 32 * 4096) return;
  int b = i >> 12, d = i & 4095;
  const h16* fp = feat + (size_t)b * 524288 + (size_t)d * 128;
  const float* al = alpha + b * 128;
  float s1 = 0.f, s2 = 0.f;
  for (int t = 0; t < 128; ++t) { float f = (float)fp[t]; float a = al[t]; s1 += a * f; s2 += a * f * f; }
  mu[i] = s1; m2[i] = s2;
}
__global__ void k_embin(h16* embin, const float* mu, const float* m2) {
  int i = blockIdx.x * blockDim.x + threadIdx.x;
  if (i >= 32 * 4096) return;
  int b = i >> 12, d = i & 4095;
  float muv = mu[i], s2 = m2[i] - muv * muv;
  float sg = sqrtf(fmaxf(s2, 1e-5f));
  embin[(size_t)b * 8192 + d] = (h16)muv;
  embin[(size_t)b * 8192 + 4096 + d] = (h16)sg;
}
__global__ void k_head(float* out, const float* emb, const float* Wc, const float* bc) {
  int i = threadIdx.x;
  if (i >= 64) return;
  int b = i >> 1, j = i & 1;
  float acc = bc[j];
  for (int k = 0; k < 256; ++k) acc += emb[b * 256 + k] * Wc[k * 2 + j];
  out[b * 2 + j] = acc;
}

// ---------------------------------------------------------------------------
// Backward (input-gradient of L = sum(out1 - out0))
// ---------------------------------------------------------------------------
__global__ void k_dembin(float* de, const float* We, const float* Wc) {  // batch-independent
  int k = blockIdx.x * blockDim.x + threadIdx.x;
  if (k >= 8192) return;
  float s = 0.f;
  for (int j = 0; j < 256; ++j) s += We[(size_t)k * 256 + j] * (Wc[2 * j + 1] - Wc[2 * j]);
  de[k] = s;
}
__global__ void k_dmug(float* g, float* dmu, const float* de, const float* mu,
                       const float* m2) {
  int i = blockIdx.x * blockDim.x + threadIdx.x;
  if (i >= 32 * 4096) return;
  int d = i & 4095;
  float muv = mu[i], s2 = m2[i] - muv * muv;
  float sg = sqrtf(fmaxf(s2, 1e-5f));
  float gg = (s2 > 1e-5f) ? de[4096 + d] * (0.5f / sg) : 0.f;
  g[i] = gg;
  dmu[i] = de[d] - 2.f * muv * gg;
}
__global__ void k_dalpha(float* dalpha, const float* dmu, const float* g,
                         const h16* feat) {
  int b = blockIdx.x >> 7, t = blockIdx.x & 127;
  __shared__ float red[256];
  const h16* fb = feat + (size_t)b * 524288 + t;
  float s = 0.f;
  for (int d = threadIdx.x; d < 4096; d += 256) {
    float f = (float)fb[(size_t)d * 128];
    s += dmu[b * 4096 + d] * f + g[b * 4096 + d] * f * f;
  }
  red[threadIdx.x] = s; __syncthreads();
  for (int w = 128; w; w >>= 1) { if (threadIdx.x < w) red[threadIdx.x] += red[threadIdx.x + w]; __syncthreads(); }
  if (!threadIdx.x) dalpha[blockIdx.x] = red[0];
}
__global__ void k_dsoft(float* ds, const float* alpha, const float* dalpha) {
  int b = blockIdx.x, t = threadIdx.x;
  __shared__ float sm[128];
  float a = alpha[b * 128 + t], da = dalpha[b * 128 + t];
  sm[t] = a * da; __syncthreads();
  for (int w = 64; w; w >>= 1) { if (t < w) sm[t] += sm[t + w]; __syncthreads(); }
  ds[b * 128 + t] = a * (da - sm[0]);
}
__global__ void k_dpre(h16* dpreT, const h16* att_b, const float* ds_b,
                       const float* va) {  // dpreT[t][h]
  int i = blockIdx.x * blockDim.x + threadIdx.x;
  if (i >= 128 * 128) return;
  int h = i >> 7, t = i & 127;
  float a = (float)att_b[i];
  dpreT[t * 128 + h] = (h16)(va[h] * ds_b[t] * (1.f - a * a));
}
__global__ void k_dfeat_fix(h16* dh3, const float* dfeat, const h16* feat_b,
                            const float* alpha_b, const float* dmu_b, const float* g_b) {
  int i = blockIdx.x * blockDim.x + threadIdx.x;
  if (i >= 4096 * 128) return;
  int d = i >> 7, t = i & 127;
  float f = (float)feat_b[i];
  float v = dfeat[i] + alpha_b[t] * (dmu_b[d] + 2.f * f * g_b[d]);
  dh3[i] = (h16)((f > 0.f) ? v : 0.f);
}
// Direct backward-data for a stride-2 3x3 conv (avg ~2.25 valid taps per pixel).
__global__ void k_conv_bwd(h16* outH, float* outF, const h16* __restrict__ dy,
                           const float* __restrict__ Wt, int Ci, int H, int W, int Co,
                           int Ho, int Wo, int maskMode, const h16* maskH,
                           const unsigned* maskBits) {
  int idx = blockIdx.x * blockDim.x + threadIdx.x;
  if (idx >= Ci * H * W) return;
  int ci = idx / (H * W), rem = idx % (H * W);
  int y = rem / W, x = rem % W;
  float acc = 0.f;
  for (int ky = 0; ky < 3; ++ky) {
    int ty = y - ky;
    if (ty < 0 || (ty & 1)) continue;
    int oy = ty >> 1;
    if (oy >= Ho) continue;
    for (int kx = 0; kx < 3; ++kx) {
      int tx = x - kx;
      if (tx < 0 || (tx & 1)) continue;
      int ox = tx >> 1;
      if (ox >= Wo) continue;
      int woff = ci * 9 + ky * 3 + kx;
      const h16* dyp = dy + (size_t)oy * Wo + ox;
      for (int co = 0; co < Co; ++co)
        acc += Wt[(size_t)co * Ci * 9 + woff] * (float)dyp[(size_t)co * Ho * Wo];
    }
  }
  if (maskMode == 1) { if (!((float)maskH[idx] > 0.f)) acc = 0.f; }
  else if (maskMode == 2) { if (!((maskBits[idx >> 5] >> (idx & 31)) & 1u)) acc = 0.f; }
  if (outF) outF[idx] = acc;
  else outH[idx] = (h16)acc;
}

// ---------------------------------------------------------------------------
// fam reductions, argsorts, RFM masking
// ---------------------------------------------------------------------------
__global__ void k_cmask(float* cm, const float* dx) {  // max over T of |dx|
  int b = blockIdx.x >> 7, f = blockIdx.x & 127;
  __shared__ float red[256];
  const float* p = dx + (size_t)b * 131072 + (size_t)f * 1024;
  float m = 0.f;
  for (int t = threadIdx.x; t < 1024; t += 256) m = fmaxf(m, fabsf(p[t]));
  red[threadIdx.x] = m; __syncthreads();
  for (int w = 128; w; w >>= 1) { if (threadIdx.x < w) red[threadIdx.x] = fmaxf(red[threadIdx.x], red[threadIdx.x + w]); __syncthreads(); }
  if (!threadIdx.x) cm[blockIdx.x] = red[0];
}
__global__ void k_tmask(float* tm, const float* dx) {  // max over F of |dx|
  int i = blockIdx.x * blockDim.x + threadIdx.x;
  if (i >= 32 * 1024) return;
  int b = i >> 10, t = i & 1023;
  float m = 0.f;
  const float* p = dx + (size_t)b * 131072 + t;
  for (int f = 0; f < 128; ++f) m = fmaxf(m, fabsf(p[f * 1024]));
  tm[i] = m;
}
__global__ void k_argsortF(int* sc, const float* cm) {  // stable descending, n=128
  int b = blockIdx.x, t = threadIdx.x;
  __shared__ float v[128];
  v[t] = cm[b * 128 + t]; __syncthreads();
  float mv = v[t]; int r = 0;
  for (int k = 0; k < 128; ++k) r += (v[k] > mv) || (v[k] == mv && k < t);
  sc[b * 128 + r] = t;
}
__global__ void k_argsortT(int* st, const float* tm) {  // n=1024, keep top-128 ranks
  int b = blockIdx.x, t = threadIdx.x;
  __shared__ float v[1024];
  v[t] = tm[b * 1024 + t]; __syncthreads();
  float mv = v[t]; int r = 0;
  for (int k = 0; k < 1024; ++k) r += (v[k] > mv) || (v[k] == mv && k < t);
  if (r < 128) st[b * 128 + r] = t;
}
__global__ void k_means(float* meanF, float* meanT, const float* x) {
  int b = blockIdx.x, t = threadIdx.x;  // 1024 threads
  const float* xb = x + (size_t)b * 131072;
  { float s = 0.f; for (int f = 0; f < 128; ++f) s += xb[f * 1024 + t]; meanF[b * 1024 + t] = s * (1.f / 128.f); }
  if (t < 128) { float s = 0.f; const float* r = xb + t * 1024; for (int k = 0; k < 1024; ++k) s += r[k]; meanT[b * 128 + t] = s * (1.f / 1024.f); }
}
// Sequential 128-step scan; code: 0=keep x, 1=row fill (meanF), 2=col fill (meanT)
__global__ void k_rfm(unsigned char* codeAll, const int* sc, const int* st,
                      const int* mask_t, const int* s_t) {
  int b = blockIdx.x;
  unsigned char* code = codeAll + (size_t)b * 131072;
  __shared__ int sh_act, sh_pc, sh_left, sh_w, sh_cnt;
  for (int j = threadIdx.x; j < 131072; j += blockDim.x) code[j] = 0;
  if (threadIdx.x == 0) sh_cnt = 0;
  __threadfence_block(); __syncthreads();
  for (int i = 0; i < 128; ++i) {
    if (threadIdx.x == 0) {
      int pc = sc[b * 128 + i], pt = st[b * 128 + i];
      int mt = mask_t[b * 128 + i], ss = s_t[b * 128 + i];
      int act = (code[pc * 1024 + pt] == 0) && (sh_cnt < 3);
      int left = pt - ss; if (left < 0) left = 0;
      int right = pt + (mt - ss); if (right > 1024) right = 1024;
      if (right < left) right = left;
      sh_act = act; sh_pc = pc; sh_left = left; sh_w = right - left;
      sh_cnt += act;
    }
    __syncthreads();
    if (sh_act)
      for (int t = threadIdx.x; t < 1024; t += blockDim.x) code[sh_pc * 1024 + t] = 1;
    __threadfence_block(); __syncthreads();
    if (sh_act && sh_w > 0) {
      int tot = 128 * sh_w;
      for (int j = threadIdx.x; j < tot; j += blockDim.x)
        code[(j / sh_w) * 1024 + sh_left + (j % sh_w)] = 2;
    }
    __threadfence_block(); __syncthreads();
  }
}
__global__ void k_apply(float* xm, const float* x, const unsigned char* code,
                        const float* meanF, const float* meanT, const int* is_train) {
  int i = blockIdx.x * blockDim.x + threadIdx.x;
  if (i >= 32 * 131072) return;
  if (!is_train[0]) { xm[i] = x[i]; return; }
  int b = i >> 17, r = i & 131071, f = r >> 10, t = r & 1023;
  unsigned char c = code[i];
  xm[i] = (c == 0) ? x[i] : (c == 1 ? meanF[b * 1024 + t] : meanT[b * 128 + f]);
}

// ---------------------------------------------------------------------------
// Host orchestration
// ---------------------------------------------------------------------------
extern "C" void kernel_launch(void* const* d_in, const int* in_sizes, int n_in,
                              void* d_out, int out_size, void* d_ws, size_t ws_size,
                              hipStream_t stream) {
  (void)in_sizes; (void)n_in; (void)out_size; (void)ws_size;
  const float* xp  = (const float*)d_in[0];
  const float* W1p = (const float*)d_in[1];  const float* b1p = (const float*)d_in[2];
  const float* W2p = (const float*)d_in[3];  const float* b2p = (const float*)d_in[4];
  const float* W3p = (const float*)d_in[5];  const float* b3p = (const float*)d_in[6];
  const float* Wap = (const float*)d_in[7];  const float* bap = (const float*)d_in[8];
  const float* vap = (const float*)d_in[9];
  const float* Wep = (const float*)d_in[10]; const float* bep = (const float*)d_in[11];
  const float* Wcp = (const float*)d_in[12]; const float* bcp = (const float*)d_in[13];
  const int* mtp   = (const int*)d_in[14];   const int* stp   = (const int*)d_in[15];
  const int* istr  = (const int*)d_in[16];
  float* outp = (float*)d_out;

  char* WS = (char*)d_ws;  size_t off = 0;
  auto alloc = [&](size_t n) { void* p = WS + off; off += (n + 255) & ~(size_t)255; return p; };
  // Workspace plan (~180 MB total)
  h16* w1f   = (h16*)alloc(64 * 32 * 2);
  h16* w2f   = (h16*)alloc(128 * 576 * 2);
  h16* w3f   = (h16*)alloc(256 * 1152 * 2);
  h16* waf   = (h16*)alloc(128 * 4096 * 2);
  h16* waT   = (h16*)alloc(4096 * 128 * 2);
  h16* weT   = (h16*)alloc(256 * 8192 * 2);
  h16* colbuf= (h16*)alloc((size_t)8192 * 576 * 2);       // max col buffer (conv2)
  h16* h1b   = (h16*)alloc((size_t)64 * 32768 * 2);       // per-batch
  h16* h2    = (h16*)alloc((size_t)32 * 128 * 8192 * 2);  // full
  h16* h3    = (h16*)alloc((size_t)32 * 256 * 2048 * 2);  // full (== feat)
  h16* att   = (h16*)alloc((size_t)32 * 128 * 128 * 2);
  float* sbuf  = (float*)alloc(32 * 128 * 4);
  float* alpha = (float*)alloc(32 * 128 * 4);
  float* mu    = (float*)alloc(32 * 4096 * 4);
  float* m2    = (float*)alloc(32 * 4096 * 4);
  unsigned* mask1 = (unsigned*)alloc((size_t)32 * 65536 * 4);
  float* dembin = (float*)alloc(8192 * 4);
  float* gbuf   = (float*)alloc(32 * 4096 * 4);
  float* dmu    = (float*)alloc(32 * 4096 * 4);
  float* dalpha = (float*)alloc(32 * 128 * 4);
  float* dsb    = (float*)alloc(32 * 128 * 4);
  h16* dpreT  = (h16*)alloc(128 * 128 * 2);
  float* dfeatb = (float*)alloc((size_t)4096 * 128 * 4);
  h16* dh3b   = (h16*)alloc((size_t)256 * 2048 * 2);
  h16* dh2b   = (h16*)alloc((size_t)128 * 8192 * 2);
  h16* dh1b   = (h16*)alloc((size_t)64 * 32768 * 2);
  float* dx   = (float*)alloc((size_t)32 * 131072 * 4);
  float* cmask = (float*)alloc(32 * 128 * 4);
  float* tmask = (float*)alloc(32 * 1024 * 4);
  int* scb = (int*)alloc(32 * 128 * 4);
  int* stb = (int*)alloc(32 * 128 * 4);
  float* meanF = (float*)alloc(32 * 1024 * 4);
  float* meanT = (float*)alloc(32 * 128 * 4);
  unsigned char* code = (unsigned char*)alloc((size_t)32 * 131072);
  float* xm = (float*)alloc((size_t)32 * 131072 * 4);
  h16* embin = (h16*)alloc((size_t)64 * 8192 * 2);  // padded to 64 rows (M tile = 64)
  float* emb = (float*)alloc(64 * 256 * 4);         // padded to 64 rows

  auto gemm = [&](const h16* A, const h16* BT, int M, int N, int K, const float* bias,
                  int act, float* Cf, h16* Ch, int ldc) {
    dim3 gr(N / 64, M / 64);
    k_gemm<<<gr, 128, 0, stream>>>(A, BT, M, N, K, bias, act, Cf, Ch, ldc);
  };
  auto blocks = [](int n) { return (n + 255) / 256; };

  // --- weight prep (f16 A-operands, pre-transposed B-operands) ---
  k_w1pack<<<blocks(2048), 256, 0, stream>>>(w1f, W1p);
  k_cvt_h<<<blocks(73728), 256, 0, stream>>>(w2f, W2p, 73728);
  k_cvt_h<<<blocks(294912), 256, 0, stream>>>(w3f, W3p, 294912);
  k_cvt_h<<<blocks(524288), 256, 0, stream>>>(waf, Wap, 524288);
  k_transpose_f32h<<<blocks(2097152), 256, 0, stream>>>(weT, Wep, 256, 8192);
  k_transpose_f32h<<<blocks(524288), 256, 0, stream>>>(waT, Wap, 4096, 128);

  auto forward = [&](const float* xin, bool full) {
    for (int b = 0; b < 32; ++b) {
      k_im2col_f32<<<blocks(1048576), 256, 0, stream>>>(colbuf, xin + (size_t)b * 131072,
                                                        1, 128, 1024, 64, 512, 32);
      gemm(w1f, colbuf, 64, 32768, 32, b1p, 1, nullptr, h1b, 32768);
      if (!full) k_mask1<<<blocks(65536), 256, 0, stream>>>(mask1 + (size_t)b * 65536, h1b, 65536);
      k_im2col_h<<<blocks(4718592), 256, 0, stream>>>(colbuf, h1b, 64, 64, 512, 32, 256, 576);
      gemm(w2f, colbuf, 128, 8192, 576, b2p, 1, nullptr, h2 + (size_t)b * 1048576, 8192);
      k_im2col_h<<<blocks(2359296), 256, 0, stream>>>(colbuf, h2 + (size_t)b * 1048576,
                                                      128, 32, 256, 16, 128, 1152);
      gemm(w3f, colbuf, 256, 2048, 1152, b3p, 1, nullptr, h3 + (size_t)b * 524288, 2048);
      k_transpose_h<<<blocks(524288), 256, 0, stream>>>(colbuf, h3 + (size_t)b * 524288, 128, 4096);
      gemm(waf, colbuf, 128, 128, 4096, bap, 2, nullptr, att + (size_t)b * 16384, 128);
    }
    k_score<<<blocks(4096), 256, 0, stream>>>(sbuf, att, vap);
    k_softmax<<<32, 128, 0, stream>>>(alpha, sbuf);
    k_pool<<<blocks(131072), 256, 0, stream>>>(mu, m2, alpha, h3);
    if (full) {
      k_zero_h<<<blocks(262144), 256, 0, stream>>>(embin + (size_t)32 * 8192, 262144);
      k_embin<<<blocks(131072), 256, 0, stream>>>(embin, mu, m2);
      gemm(embin, weT, 64, 256, 8192, bep, 0, emb, nullptr, 256);
      k_head<<<1, 64, 0, stream>>>(outp, emb, Wcp, bcp);
    }
  };

  // ---- pass 1: forward on x (stats needed for the gradient) ----
  forward(xp, false);

  // ---- backward to d_x ----
  k_dembin<<<blocks(8192), 256, 0, stream>>>(dembin, Wep, Wcp);
  k_dmug<<<blocks(131072), 256, 0, stream>>>(gbuf, dmu, dembin, mu, m2);
  k_dalpha<<<4096, 256, 0, stream>>>(dalpha, dmu, gbuf, h3);
  k_dsoft<<<32, 128, 0, stream>>>(dsb, alpha, dalpha);
  for (int b = 0; b < 32; ++b) {
    k_dpre<<<blocks(16384), 256, 0, stream>>>(dpreT, att + (size_t)b * 16384,
                                              dsb + b * 128, vap);
    gemm(waT, dpreT, 4096, 128, 128, nullptr, 0, dfeatb, nullptr, 128);
    k_dfeat_fix<<<blocks(524288), 256, 0, stream>>>(dh3b, dfeatb, h3 + (size_t)b * 524288,
                                                    alpha + b * 128, dmu + b * 4096,
                                                    gbuf + b * 4096);
    k_conv_bwd<<<blocks(1048576), 256, 0, stream>>>(dh2b, nullptr, dh3b, W3p, 128, 32, 256,
                                                    256, 16, 128, 1,
                                                    h2 + (size_t)b * 1048576, nullptr);
    k_conv_bwd<<<blocks(2097152), 256, 0, stream>>>(dh1b, nullptr, dh2b, W2p, 64, 64, 512,
                                                    128, 32, 256, 2, nullptr,
                                                    mask1 + (size_t)b * 65536);
    k_conv_bwd<<<blocks(131072), 256, 0, stream>>>(nullptr, dx + (size_t)b * 131072, dh1b,
                                                   W1p, 1, 128, 1024, 64, 64, 512, 0,
                                                   nullptr, nullptr);
  }

  // ---- RFM masking ----
  k_cmask<<<4096, 256, 0, stream>>>(cmask, dx);
  k_tmask<<<blocks(32768), 256, 0, stream>>>(tmask, dx);
  k_argsortF<<<32, 128, 0, stream>>>(scb, cmask);
  k_argsortT<<<32, 1024, 0, stream>>>(stb, tmask);
  k_means<<<32, 1024, 0, stream>>>(meanF, meanT, xp);
  k_rfm<<<32, 256, 0, stream>>>(code, scb, stb, mtp, stp);
  k_apply<<<blocks(4194304), 256, 0, stream>>>(xm, xp, code, meanF, meanT, istr);

  // ---- pass 2: full model on masked x -> d_out ----
  forward(xm, true);
}
